// PositionAttention_42331197669735
// MI455X (gfx1250) — matrix operand order repaired and checked by reading, not compiled, not run on hardware
//
#include <hip/hip_runtime.h>
#include <cstddef>

// ---------------------------------------------------------------------------
// MI455X (gfx1250) position-attention, bf16 WMMA + async global->LDS pipeline.
//   P1: stage stacked W ([Wq;Wk;Wv]) as bf16 [640][512]
//   P2: stage x transposed as bf16 xT[n][p][c]   (GEMM-B async-friendly)
//   K1: Y = W x xf via WMMA; Q/K stored transposed Yqk[n][p][d0..127],
//       V stored Yv[n][c][p]  (all tiles row-contiguous for async copies)
//   K2: softmax row stats (m,l) of Q^T K, S via WMMA (never materialized)
//   K3: O = (exp(Q^T K - m)/l) * V^T via WMMA; out = gama*O^T + x
// All hot-loop LDS tile fills use global_load_async_to_lds_b128 (ASYNCcnt).
// ---------------------------------------------------------------------------

typedef __attribute__((ext_vector_type(16))) __bf16        v16bf;
typedef __attribute__((ext_vector_type(8)))  float         v8f;
typedef __attribute__((ext_vector_type(4)))  unsigned int  u32x4;

union FragU { v16bf v; u32x4 q[2]; };

__device__ __forceinline__ unsigned short f2bf(float f) {
  unsigned int u = __float_as_uint(f);
  u += 0x7fffu + ((u >> 16) & 1u);          // round-to-nearest-even
  return (unsigned short)(u >> 16);
}

// Generic pointer -> 32-bit LDS offset (addrspace(3) value) for async VDST.
typedef __attribute__((address_space(3))) unsigned char lds_byte;
__device__ __forceinline__ unsigned lds_off_of(const void* p) {
  return (unsigned)(unsigned long long)(lds_byte*)p;
}

// 16-byte async copy global -> LDS (GV mode: vaddr64 + off). ASYNCcnt-tracked.
__device__ __forceinline__ void async_cp16(unsigned lds_off, const void* gptr) {
  asm volatile("global_load_async_to_lds_b128 %0, %1, off"
               :: "v"(lds_off), "v"((unsigned long long)(size_t)gptr)
               : "memory");
}
__device__ __forceinline__ void wait_async0() {
  asm volatile("s_wait_asynccnt 0" ::: "memory");
}

__device__ __forceinline__ v16bf ld_frag(const unsigned short* p0,
                                         const unsigned short* p1) {
  FragU f;
  f.q[0] = *(const u32x4*)p0;
  f.q[1] = *(const u32x4*)p1;
  return f.v;
}

__device__ __forceinline__ v8f zero8() { v8f z = {}; return z; }

__device__ __forceinline__ v8f wmma_bf16(v16bf a, v16bf b, v8f c) {
  return __builtin_amdgcn_wmma_f32_16x16x32_bf16(false, a, false, b,
                                                 (short)0, c, false, false);
}

// ---------------------------------------------------------------------------
// Prep 1: stacked bf16 weights Wst[640][512]. Grid(640), block 128.
// Row class is uniform per block -> no divergence.
// ---------------------------------------------------------------------------
__global__ __launch_bounds__(128) void prep_w_kernel(
    const float* __restrict__ Wq, const float* __restrict__ Wk,
    const float* __restrict__ Wv, unsigned short* __restrict__ Wst) {
  const int row = blockIdx.x, t = threadIdx.x;
  const float* src;
  if (row < 64)       src = Wq + (size_t)row * 512;
  else if (row < 128) src = Wk + (size_t)(row - 64) * 512;
  else                src = Wv + (size_t)(row - 128) * 512;
#pragma unroll
  for (int i = 0; i < 4; ++i) {
    int c = i * 128 + t;
    Wst[(size_t)row * 512 + c] = f2bf(src[c]);
  }
}

// ---------------------------------------------------------------------------
// Prep 2: xT[n][p][c] = bf16(x[n][c][p]). LDS-tiled transpose.
// Grid (64 ptiles, 8 ctiles, 8 n), block 256.
// ---------------------------------------------------------------------------
__global__ __launch_bounds__(256) void prep_xt_kernel(
    const float* __restrict__ x, unsigned short* __restrict__ xT) {
  __shared__ __align__(16) unsigned short tl[64][72];
  const int t = threadIdx.x;
  const int pbase = blockIdx.x * 64, cbase = blockIdx.y * 64, n = blockIdx.z;
#pragma unroll
  for (int i = 0; i < 16; ++i) {
    int idx = i * 256 + t;
    int p = idx & 63, ci = idx >> 6;
    tl[ci][p] = f2bf(x[((size_t)n * 512 + cbase + ci) * 4096 + pbase + p]);
  }
  __syncthreads();
#pragma unroll
  for (int i = 0; i < 16; ++i) {
    int idx = i * 256 + t;
    int ci = idx & 63, p = idx >> 6;
    xT[((size_t)n * 4096 + pbase + p) * 512 + cbase + ci] = tl[ci][p];
  }
}

// ---------------------------------------------------------------------------
// Kernel 1: QKV GEMM. Grid (64 ptiles, 10 rowtiles, 8 n), block 128 (4 waves).
// Wave computes 16 rows x 64 pixels. Tiles filled with async b128 copies.
// ---------------------------------------------------------------------------
__global__ __launch_bounds__(128) void qkv_gemm_kernel(
    const unsigned short* __restrict__ Wst, const unsigned short* __restrict__ xT,
    unsigned short* __restrict__ Yqk, unsigned short* __restrict__ Yv) {
  __shared__ __align__(16) unsigned short lds_a[64][40];  // W rows x 32 K
  __shared__ __align__(16) unsigned short lds_b[64][40];  // pixels x 32 K
  const int t = threadIdx.x;
  const int lane = t & 31, w = t >> 5;
  const int grp = lane >> 4, l16 = lane & 15;
  const int pbase = blockIdx.x * 64;
  const int rbase = blockIdx.y * 64;
  const int n = blockIdx.z;
  const unsigned baseA = lds_off_of(&lds_a[0][0]);
  const unsigned baseB = lds_off_of(&lds_b[0][0]);

  v8f acc[4];
#pragma unroll
  for (int s = 0; s < 4; ++s) acc[s] = zero8();

  for (int kk = 0; kk < 16; ++kk) {
    const int k0 = kk * 32;
    __syncthreads();
#pragma unroll
    for (int i = 0; i < 2; ++i) {            // A tile: 64 rows x 64B
      int c = i * 128 + t, r = c >> 2, sub = c & 3;
      async_cp16(baseA + (unsigned)(r * 40 + sub * 8) * 2u,
                 Wst + ((size_t)(rbase + r) * 512 + k0 + sub * 8));
    }
#pragma unroll
    for (int i = 0; i < 2; ++i) {            // B tile: 64 pixel-rows x 64B
      int c = i * 128 + t, p = c >> 2, sub = c & 3;
      async_cp16(baseB + (unsigned)(p * 40 + sub * 8) * 2u,
                 xT + ((size_t)(n * 4096 + pbase + p) * 512 + k0 + sub * 8));
    }
    wait_async0();
    __syncthreads();
    // A frag: lane<16 holds K 0-7 & 16-23, lane>=16 holds K 8-15 & 24-31
    const int am = w * 16 + l16;
    const int ac = grp ? 8 : 0;
    v16bf af = ld_frag(&lds_a[am][ac], &lds_a[am][ac + 16]);
    const int bk = grp ? 8 : 0;              // B frag: lane<16 K 0-7 & 16-23, else 8-15 & 24-31
#pragma unroll
    for (int s = 0; s < 4; ++s) {
      v16bf bf = ld_frag(&lds_b[s * 16 + l16][bk], &lds_b[s * 16 + l16][bk + 16]);
      acc[s] = wmma_bf16(af, bf, acc[s]);
    }
  }
  // Store: Q/K rows transposed into Yqk[n][p][row]; V rows into Yv[n][c][p].
  if (rbase < 128) {
#pragma unroll
    for (int s = 0; s < 4; ++s)
#pragma unroll
      for (int e = 0; e < 8; ++e) {
        int row = rbase + w * 16 + e + 8 * grp;
        int p = pbase + s * 16 + l16;
        Yqk[((size_t)n * 4096 + p) * 128 + row] = f2bf(acc[s][e]);
      }
  } else {
#pragma unroll
    for (int s = 0; s < 4; ++s)
#pragma unroll
      for (int e = 0; e < 8; ++e) {
        int cch = rbase - 128 + w * 16 + e + 8 * grp;
        int p = pbase + s * 16 + l16;
        Yv[((size_t)n * 512 + cch) * 4096 + p] = f2bf(acc[s][e]);
      }
  }
}

// ---------------------------------------------------------------------------
// Kernel 2: softmax row stats. Grid (64 qtiles, 8 n), block 128.
// ---------------------------------------------------------------------------
__global__ __launch_bounds__(128) void attn_stats_kernel(
    const unsigned short* __restrict__ Yqk, float* __restrict__ mbuf,
    float* __restrict__ lbuf) {
  __shared__ __align__(16) unsigned short lds_q[64][72];  // [query][d]
  __shared__ __align__(16) unsigned short lds_k[64][72];  // [key j][d]
  const int t = threadIdx.x, lane = t & 31, w = t >> 5;
  const int grp = lane >> 4, l16 = lane & 15;
  const int qbase = blockIdx.x * 64;
  const int n = blockIdx.y;
  const unsigned baseQ = lds_off_of(&lds_q[0][0]);
  const unsigned baseK = lds_off_of(&lds_k[0][0]);

#pragma unroll
  for (int i = 0; i < 4; ++i) {              // Q tile: rows of 128B (d 0..63)
    int c = i * 128 + t, r = c >> 3, sub = c & 7;
    async_cp16(baseQ + (unsigned)(r * 72 + sub * 8) * 2u,
               Yqk + ((size_t)(n * 4096 + qbase + r) * 128 + sub * 8));
  }
  wait_async0();
  __syncthreads();
  const int am = w * 16 + l16, ac = grp ? 8 : 0, bk = grp ? 8 : 0;
  v16bf aq[2];
#pragma unroll
  for (int kd = 0; kd < 2; ++kd)
    aq[kd] = ld_frag(&lds_q[am][kd * 32 + ac], &lds_q[am][kd * 32 + ac + 16]);

  float m_run[8], l_run[8];
#pragma unroll
  for (int e = 0; e < 8; ++e) { m_run[e] = -3.0e38f; l_run[e] = 0.f; }

  for (int jt = 0; jt < 64; ++jt) {
    __syncthreads();
#pragma unroll
    for (int i = 0; i < 4; ++i) {            // K tile: rows of 128B (d 64..127)
      int c = i * 128 + t, r = c >> 3, sub = c & 7;
      async_cp16(baseK + (unsigned)(r * 72 + sub * 8) * 2u,
                 Yqk + ((size_t)(n * 4096 + jt * 64 + r) * 128 + 64 + sub * 8));
    }
    wait_async0();
    __syncthreads();
    v8f sa[4];
#pragma unroll
    for (int s = 0; s < 4; ++s) {
      sa[s] = zero8();
#pragma unroll
      for (int kd = 0; kd < 2; ++kd) {
        v16bf bf = ld_frag(&lds_k[s * 16 + l16][kd * 32 + bk],
                           &lds_k[s * 16 + l16][kd * 32 + bk + 16]);
        sa[s] = wmma_bf16(aq[kd], bf, sa[s]);
      }
    }
#pragma unroll
    for (int e = 0; e < 8; ++e) {            // online max/sum per query row
      float pm = fmaxf(fmaxf(sa[0][e], sa[1][e]), fmaxf(sa[2][e], sa[3][e]));
#pragma unroll
      for (int msk = 1; msk <= 8; msk <<= 1) pm = fmaxf(pm, __shfl_xor(pm, msk, 32));
      float m_new = fmaxf(m_run[e], pm);
      float ps = __expf(sa[0][e] - m_new) + __expf(sa[1][e] - m_new) +
                 __expf(sa[2][e] - m_new) + __expf(sa[3][e] - m_new);
#pragma unroll
      for (int msk = 1; msk <= 8; msk <<= 1) ps += __shfl_xor(ps, msk, 32);
      l_run[e] = l_run[e] * __expf(m_run[e] - m_new) + ps;
      m_run[e] = m_new;
    }
  }
  if (l16 == 0) {                            // lanes 0 & 16 hold rows e / e+8
#pragma unroll
    for (int e = 0; e < 8; ++e) {
      int q = qbase + w * 16 + e + 8 * grp;
      mbuf[n * 4096 + q] = m_run[e];
      lbuf[n * 4096 + q] = l_run[e];
    }
  }
}

// ---------------------------------------------------------------------------
// Kernel 3: O = softmax(S) * V^T, out = gama*O^T + x.
// Grid (8 ctiles, 64 qtiles, 8 n), block 128; wave = 16 q x 64 c.
// ---------------------------------------------------------------------------
__global__ __launch_bounds__(128) void attn_out_kernel(
    const unsigned short* __restrict__ Yqk, const unsigned short* __restrict__ Yv,
    const float* __restrict__ mbuf, const float* __restrict__ lbuf,
    const float* __restrict__ x, const float* __restrict__ gama,
    float* __restrict__ out) {
  __shared__ __align__(16) unsigned short lds_q[64][72];      // [q][d]
  __shared__ __align__(16) unsigned short lds_k[64][72];      // [j][d]
  __shared__ __align__(16) unsigned short lds_v[64][72];      // [c][j]
  __shared__ __align__(16) unsigned short lds_p[4][16][72];   // per-wave P stage
  const int t = threadIdx.x, lane = t & 31, w = t >> 5;
  const int grp = lane >> 4, l16 = lane & 15;
  const int cbase = blockIdx.x * 64;
  const int qbase = blockIdx.y * 64;
  const int n = blockIdx.z;
  const unsigned baseQ = lds_off_of(&lds_q[0][0]);
  const unsigned baseK = lds_off_of(&lds_k[0][0]);
  const unsigned baseV = lds_off_of(&lds_v[0][0]);

  float m_q[8], li_q[8];
#pragma unroll
  for (int e = 0; e < 8; ++e) {
    int q = qbase + w * 16 + e + 8 * grp;
    m_q[e] = mbuf[n * 4096 + q];
    li_q[e] = 1.0f / lbuf[n * 4096 + q];
  }

#pragma unroll
  for (int i = 0; i < 4; ++i) {              // Q tile
    int c = i * 128 + t, r = c >> 3, sub = c & 7;
    async_cp16(baseQ + (unsigned)(r * 72 + sub * 8) * 2u,
               Yqk + ((size_t)(n * 4096 + qbase + r) * 128 + sub * 8));
  }
  wait_async0();
  __syncthreads();
  const int am = w * 16 + l16, ac = grp ? 8 : 0, bk = grp ? 8 : 0;
  v16bf aq[2];
#pragma unroll
  for (int kd = 0; kd < 2; ++kd)
    aq[kd] = ld_frag(&lds_q[am][kd * 32 + ac], &lds_q[am][kd * 32 + ac + 16]);

  v8f o_acc[4];
#pragma unroll
  for (int cs = 0; cs < 4; ++cs) o_acc[cs] = zero8();

  for (int jt = 0; jt < 64; ++jt) {
    __syncthreads();
#pragma unroll
    for (int i = 0; i < 4; ++i) {            // K tile
      int c = i * 128 + t, r = c >> 3, sub = c & 7;
      async_cp16(baseK + (unsigned)(r * 72 + sub * 8) * 2u,
                 Yqk + ((size_t)(n * 4096 + jt * 64 + r) * 128 + 64 + sub * 8));
    }
#pragma unroll
    for (int i = 0; i < 4; ++i) {            // V tile (rows contiguous in j)
      int c = i * 128 + t, r = c >> 3, sub = c & 7;
      async_cp16(baseV + (unsigned)(r * 72 + sub * 8) * 2u,
                 Yv + ((size_t)(n * 512 + cbase + r) * 4096 + jt * 64 + sub * 8));
    }
    wait_async0();
    __syncthreads();
    // S = Q^T K, then P = exp(S - m)/l staged to LDS in A-fragment order.
    // (per-wave staging: LDS ops are in-order within a wave, no barrier needed)
#pragma unroll
    for (int s = 0; s < 4; ++s) {
      v8f sa = zero8();
#pragma unroll
      for (int kd = 0; kd < 2; ++kd) {
        v16bf bf = ld_frag(&lds_k[s * 16 + l16][kd * 32 + bk],
                           &lds_k[s * 16 + l16][kd * 32 + bk + 16]);
        sa = wmma_bf16(aq[kd], bf, sa);
      }
#pragma unroll
      for (int e = 0; e < 8; ++e) {
        float p = __expf(sa[e] - m_q[e]) * li_q[e];
        lds_p[w][e + 8 * grp][s * 16 + l16] = f2bf(p);   // row=query, col=j
      }
    }
    // O += P(16xJ) x V^T(Jx64): A frag from lds_p, B frag from lds_v[c][j].
#pragma unroll
    for (int jc = 0; jc < 2; ++jc) {
      v16bf pf = ld_frag(&lds_p[w][l16][jc * 32 + ac],
                         &lds_p[w][l16][jc * 32 + ac + 16]);
#pragma unroll
      for (int cs = 0; cs < 4; ++cs) {
        v16bf vf = ld_frag(&lds_v[cs * 16 + l16][jc * 32 + bk],
                           &lds_v[cs * 16 + l16][jc * 32 + bk + 16]);
        o_acc[cs] = wmma_bf16(pf, vf, o_acc[cs]);
      }
    }
  }
  const float g = gama[0];
#pragma unroll
  for (int cs = 0; cs < 4; ++cs)
#pragma unroll
    for (int e = 0; e < 8; ++e) {
      int c = cbase + cs * 16 + l16;
      int q = qbase + w * 16 + e + 8 * grp;
      size_t o = ((size_t)n * 512 + c) * 4096 + q;
      out[o] = g * o_acc[cs][e] + x[o];
    }
}

// ---------------------------------------------------------------------------
extern "C" void kernel_launch(void* const* d_in, const int* in_sizes, int n_in,
                              void* d_out, int out_size, void* d_ws, size_t ws_size,
                              hipStream_t stream) {
  const float* x    = (const float*)d_in[0];
  const float* Wq   = (const float*)d_in[1];
  const float* Wk   = (const float*)d_in[2];
  const float* Wv   = (const float*)d_in[3];
  const float* gama = (const float*)d_in[4];
  float* out = (float*)d_out;

  // Workspace layout (halves unless noted):
  //   Wst [640][512]            =    327,680
  //   xT  [8][4096][512]        = 16,777,216
  //   Yqk [8][4096][128]        =  4,194,304
  //   Yv  [8][512][4096]        = 16,777,216
  //   mbuf,lbuf float [8][4096]
  unsigned short* Wst = (unsigned short*)d_ws;
  unsigned short* xT  = Wst + (size_t)640 * 512;
  unsigned short* Yqk = xT  + (size_t)8 * 4096 * 512;
  unsigned short* Yv  = Yqk + (size_t)8 * 4096 * 128;
  float* mbuf = (float*)(Yv + (size_t)8 * 512 * 4096);
  float* lbuf = mbuf + (size_t)8 * 4096;
  (void)in_sizes; (void)n_in; (void)out_size; (void)ws_size;

  prep_w_kernel<<<dim3(640), 128, 0, stream>>>(Wq, Wk, Wv, Wst);
  prep_xt_kernel<<<dim3(64, 8, 8), 256, 0, stream>>>(x, xT);
  qkv_gemm_kernel<<<dim3(64, 10, 8), 128, 0, stream>>>(Wst, xT, Yqk, Yv);
  attn_stats_kernel<<<dim3(64, 8), 128, 0, stream>>>(Yqk, mbuf, lbuf);
  attn_out_kernel<<<dim3(8, 64, 8), 128, 0, stream>>>(Yqk, Yv, mbuf, lbuf, x, gama, out);
}
